// RGCN_25855703122643
// MI455X (gfx1250) — matrix operand order repaired and checked by reading, not compiled, not run on hardware
//
#include <hip/hip_runtime.h>
#include <hip/hip_bf16.h>

typedef __attribute__((ext_vector_type(16))) _Float16 v16h;
typedef __attribute__((ext_vector_type(8)))  _Float16 v8h;
typedef __attribute__((ext_vector_type(8)))  float    v8f;

#define NEG_SLOPE 0.2f

// ---------------------------------------------------------------------------
// Conversions
// ---------------------------------------------------------------------------
__global__ void k_f32_to_f16(const float* __restrict__ in, _Float16* __restrict__ out, int n) {
    int i = blockIdx.x * blockDim.x + threadIdx.x;
    if (i < n) out[i] = (_Float16)in[i];
}

// W1 [R,128,3,64] -> Wt [R,192,128] f16, Wt[r][h*64+o][i] = W1[r][i][h][o]
__global__ void k_conv_W1(const float* __restrict__ W, _Float16* __restrict__ Wt, int R) {
    int i = blockIdx.x * blockDim.x + threadIdx.x;
    int total = R * 192 * 128;
    if (i >= total) return;
    int r = i / (192 * 128);
    int rem = i % (192 * 128);
    int hc = rem / 128;       // h*64 + o
    int ii = rem % 128;       // input feature
    int h = hc >> 6, o = hc & 63;
    Wt[i] = (_Float16)W[((r * 128 + ii) * 3 + h) * 64 + o];
}

// W2 [R,64,1,64] -> Wt [R,64,64] f16, Wt[r][o][i] = W2[r][i][0][o]
__global__ void k_conv_W2(const float* __restrict__ W, _Float16* __restrict__ Wt, int R) {
    int i = blockIdx.x * blockDim.x + threadIdx.x;
    int total = R * 64 * 64;
    if (i >= total) return;
    int r = i / 4096;
    int rem = i % 4096;
    int o = rem / 64;
    int ii = rem % 64;
    Wt[i] = (_Float16)W[(r * 64 + ii) * 64 + o];
}

__global__ void k_fill_zero(unsigned* __restrict__ p, int n) {
    int i = blockIdx.x * blockDim.x + threadIdx.x;
    if (i < n) p[i] = 0u;
}

// ---------------------------------------------------------------------------
// WMMA GEMM: C[N,Kout] = A[N,K] * Bt[Kout,K]^T   (A,Bt row-major f16, C f32)
// grid = (N/16, Kout/(16*WAVES)), block = 32*WAVES. N%16==0, K%32==0, Kout%(16*WAVES)==0.
// ---------------------------------------------------------------------------
template <int WAVES>
__global__ void wmma_gemm_f16(const _Float16* __restrict__ A,
                              const _Float16* __restrict__ Bt,
                              float* __restrict__ C,
                              int K, int Kout) {
    int lane = threadIdx.x & 31;
    int wave = threadIdx.x >> 5;
    int row0 = blockIdx.x * 16;
    int col0 = (blockIdx.y * WAVES + wave) * 16;
    int m  = lane & 15;
    int hi = lane >> 4;   // 0 or 1

    const _Float16* arow = A  + (size_t)(row0 + m) * K;
    const _Float16* brow = Bt + (size_t)(col0 + m) * K;

    v8f acc = {};
    for (int kb = 0; kb < K; kb += 32) {
        v16h a, b;
        // A 16x32 f16: lanes 0-15 hold K 0..15, lanes 16-31 hold K 16..31 (contiguous, matches B)
        *((v8h*)&a)     = *(const v8h*)(arow + kb + hi * 16);
        *((v8h*)&a + 1) = *(const v8h*)(arow + kb + hi * 16 + 8);
        // B 32x16 f16: lanes 0-15 hold K 0..15, lanes 16-31 hold K 16..31 (contiguous in Bt row)
        b = *(const v16h*)(brow + kb + hi * 16);
        acc = __builtin_amdgcn_wmma_f32_16x16x32_f16(
            /*neg_a=*/false, a, /*neg_b=*/false, b,
            /*c_mod=*/(short)0, acc, /*reuse_a=*/false, /*reuse_b=*/false);
    }
#pragma unroll
    for (int rr = 0; rr < 8; ++rr) {
        C[(size_t)(row0 + rr + hi * 8) * Kout + col0 + m] = acc[rr];
    }
}

// ---------------------------------------------------------------------------
// Attention logits: el[n,h] = sum_d h[n,h*D+d]*al[h*D+d] ; likewise er.
// ---------------------------------------------------------------------------
__global__ void k_attn_logits(const float* __restrict__ h,
                              const float* __restrict__ al, const float* __restrict__ ar,
                              float* __restrict__ el, float* __restrict__ er,
                              int N, int H, int D) {
    int i = blockIdx.x * blockDim.x + threadIdx.x;
    if (i >= N * H) return;
    int n = i / H, hh = i % H;
    const float* hp  = h + (size_t)n * (H * D) + hh * D;
    const float* alp = al + hh * D;
    const float* arp = ar + hh * D;
    float sl = 0.f, sr = 0.f;
    for (int d = 0; d < D; ++d) { float v = hp[d]; sl += v * alp[d]; sr += v * arp[d]; }
    el[i] = sl; er[i] = sr;
}

// monotone float<->uint encoding for atomic max on floats
__device__ __forceinline__ unsigned enc_f(float f) {
    unsigned b = __float_as_uint(f);
    return (b & 0x80000000u) ? ~b : (b | 0x80000000u);
}
__device__ __forceinline__ float dec_f(unsigned u) {
    return __uint_as_float((u & 0x80000000u) ? (u ^ 0x80000000u) : ~u);
}
__device__ __forceinline__ float lrelu(float v) { return v >= 0.f ? v : NEG_SLOPE * v; }

__global__ void k_edge_max(const int* __restrict__ src, const int* __restrict__ dst,
                           const float* __restrict__ el, const float* __restrict__ er,
                           unsigned* __restrict__ mEnc, int E, int H) {
    int e = blockIdx.x * blockDim.x + threadIdx.x;
    if (e >= E) return;
    int s = src[e], d = dst[e];
    for (int h = 0; h < H; ++h) {
        float v = lrelu(el[s * H + h] + er[d * H + h]);
        atomicMax(&mEnc[d * H + h], enc_f(v));
    }
}

// sentinel 0 => empty segment => m = 0 (matches reference isfinite fix-up)
__global__ void k_fixup_m(const unsigned* __restrict__ mEnc, float* __restrict__ m, int n) {
    int i = blockIdx.x * blockDim.x + threadIdx.x;
    if (i >= n) return;
    unsigned u = mEnc[i];
    m[i] = (u == 0u) ? 0.f : dec_f(u);
}

__global__ void k_edge_exp(const int* __restrict__ src, const int* __restrict__ dst,
                           const float* __restrict__ el, const float* __restrict__ er,
                           const float* __restrict__ m, float* __restrict__ pbuf,
                           float* __restrict__ denom, int E, int H) {
    int e = blockIdx.x * blockDim.x + threadIdx.x;
    if (e >= E) return;
    int s = src[e], d = dst[e];
    for (int h = 0; h < H; ++h) {
        float v = lrelu(el[s * H + h] + er[d * H + h]);
        float p = __expf(v - m[d * H + h]);
        pbuf[(size_t)e * H + h] = p;
        atomicAdd(&denom[d * H + h], p);
    }
}

// one 32-lane group per edge; lanes cover D=64 with two coalesced atomics per head
__global__ void k_edge_scatter(const int* __restrict__ src, const int* __restrict__ dst,
                               const float* __restrict__ pbuf, const float* __restrict__ denom,
                               const float* __restrict__ hfeat, float* __restrict__ agg,
                               int E, int H) {
    int lane = threadIdx.x & 31;
    int e = blockIdx.x * (blockDim.x >> 5) + (threadIdx.x >> 5);
    if (e >= E) return;
    int s = src[e], d = dst[e];
    int HD = H * 64;
    for (int h = 0; h < H; ++h) {
        float alpha = pbuf[(size_t)e * H + h] / (denom[d * H + h] + 1e-9f);
        const float* hs = hfeat + (size_t)s * HD + h * 64;
        float* ag = agg + (size_t)d * HD + h * 64;
        atomicAdd(&ag[lane],      alpha * hs[lane]);
        atomicAdd(&ag[lane + 32], alpha * hs[lane + 32]);
    }
}

// mean over 3 heads: out[n,d] = (agg[n,d] + agg[n,64+d] + agg[n,128+d]) / 3
__global__ void k_head_mean3(const float* __restrict__ agg, float* __restrict__ out, int N) {
    int i = blockIdx.x * blockDim.x + threadIdx.x;
    if (i >= N * 64) return;
    int n = i >> 6, d = i & 63;
    const float* a = agg + (size_t)n * 192;
    out[i] = (a[d] + a[64 + d] + a[128 + d]) * (1.f / 3.f);
}

// ---------------------------------------------------------------------------
// Semantic attention: s[id] = q . tanh(feat[id,:64] @ semW[64,128] + b)
// one 128-thread block per (r,n), LDS tree reduction
// ---------------------------------------------------------------------------
__global__ void k_sem_score(const float* __restrict__ feats, const float* __restrict__ semW,
                            const float* __restrict__ semb, const float* __restrict__ semq,
                            float* __restrict__ sout) {
    int id = blockIdx.x;
    int t  = threadIdx.x;                 // 0..127
    const float* f = feats + (size_t)id * 64;
    float acc = semb[t];
    for (int d = 0; d < 64; ++d) acc += f[d] * semW[d * 128 + t];
    acc = tanhf(acc) * semq[t];
    __shared__ float red[128];
    red[t] = acc;
    __syncthreads();
    for (int s2 = 64; s2 > 0; s2 >>= 1) {
        if (t < s2) red[t] += red[t + s2];
        __syncthreads();
    }
    if (t == 0) sout[id] = red[0];
}

// softmax over R=4 relations, combine, relu, emit f16 (layer-1 output)
__global__ void k_combine_relu_f16(const float* __restrict__ outs, const float* __restrict__ s,
                                   _Float16* __restrict__ out16, int N) {
    int i = blockIdx.x * blockDim.x + threadIdx.x;
    if (i >= N * 64) return;
    int n = i >> 6;
    float s0 = s[n], s1 = s[N + n], s2 = s[2 * N + n], s3 = s[3 * N + n];
    float mx = fmaxf(fmaxf(s0, s1), fmaxf(s2, s3));
    float e0 = __expf(s0 - mx), e1 = __expf(s1 - mx), e2 = __expf(s2 - mx), e3 = __expf(s3 - mx);
    float inv = 1.f / (e0 + e1 + e2 + e3);
    size_t NN = (size_t)N * 64;
    float v = (e0 * outs[i] + e1 * outs[NN + i] + e2 * outs[2 * NN + i] + e3 * outs[3 * NN + i]) * inv;
    out16[i] = (_Float16)(v > 0.f ? v : 0.f);
}

// final combine, no relu, f32 output
__global__ void k_combine_out(const float* __restrict__ outs, const float* __restrict__ s,
                              float* __restrict__ out, int N) {
    int i = blockIdx.x * blockDim.x + threadIdx.x;
    if (i >= N * 64) return;
    int n = i >> 6;
    float s0 = s[n], s1 = s[N + n], s2 = s[2 * N + n], s3 = s[3 * N + n];
    float mx = fmaxf(fmaxf(s0, s1), fmaxf(s2, s3));
    float e0 = __expf(s0 - mx), e1 = __expf(s1 - mx), e2 = __expf(s2 - mx), e3 = __expf(s3 - mx);
    float inv = 1.f / (e0 + e1 + e2 + e3);
    size_t NN = (size_t)N * 64;
    out[i] = (e0 * outs[i] + e1 * outs[NN + i] + e2 * outs[2 * NN + i] + e3 * outs[3 * NN + i]) * inv;
}

// ---------------------------------------------------------------------------
// Host orchestration
// ---------------------------------------------------------------------------
extern "C" void kernel_launch(void* const* d_in, const int* in_sizes, int n_in,
                              void* d_out, int out_size, void* d_ws, size_t ws_size,
                              hipStream_t stream) {
    const int N = 50000, R = 4, E = 1000000;

    const float* x       = (const float*)d_in[0];
    const int*   esrc    = (const int*)d_in[1];     // [R,E]
    const int*   edst    = (const int*)d_in[2];     // [R,E]
    const float* W1      = (const float*)d_in[3];   // [R,128,3,64]
    const float* al1     = (const float*)d_in[4];   // [R,3,64]
    const float* ar1     = (const float*)d_in[5];
    const float* semW1   = (const float*)d_in[6];   // [64,128]
    const float* semb1   = (const float*)d_in[7];
    const float* semq1   = (const float*)d_in[8];
    const float* W2      = (const float*)d_in[9];   // [R,64,1,64]
    const float* al2     = (const float*)d_in[10];  // [R,1,64]
    const float* ar2     = (const float*)d_in[11];
    const float* semW2   = (const float*)d_in[12];
    const float* semb2   = (const float*)d_in[13];
    const float* semq2   = (const float*)d_in[14];
    float* out = (float*)d_out;

    // workspace sub-allocation (256B aligned)
    char* wsb = (char*)d_ws;
    size_t off = 0;
    auto alloc = [&](size_t bytes) -> void* {
        void* p = wsb + off;
        off = (off + bytes + 255) & ~(size_t)255;
        return p;
    };
    _Float16* x16    = (_Float16*)alloc((size_t)N * 128 * 2);
    _Float16* W1t    = (_Float16*)alloc((size_t)R * 192 * 128 * 2);
    _Float16* W2t    = (_Float16*)alloc((size_t)R * 64 * 64 * 2);
    float*    out1   = (float*)   alloc((size_t)R * N * 64 * 4);
    float*    out2   = (float*)   alloc((size_t)R * N * 64 * 4);
    _Float16* hl16   = (_Float16*)alloc((size_t)N * 64 * 2);
    float*    s1     = (float*)   alloc((size_t)R * N * 4);
    float*    s2     = (float*)   alloc((size_t)R * N * 4);
    float*    hproj  = (float*)   alloc((size_t)N * 192 * 4);  // layer1 feats (layer2 reuses)
    float*    el     = (float*)   alloc((size_t)N * 3 * 4);
    float*    er     = (float*)   alloc((size_t)N * 3 * 4);
    unsigned* mEnc   = (unsigned*)alloc((size_t)N * 3 * 4);
    float*    mDec   = (float*)   alloc((size_t)N * 3 * 4);
    float*    denom  = (float*)   alloc((size_t)N * 3 * 4);
    float*    pbuf   = (float*)   alloc((size_t)E * 3 * 4);
    float*    agg    = (float*)   alloc((size_t)N * 192 * 4);

    const int TB = 256;
    auto blk = [](int n, int t) { return (n + t - 1) / t; };

    // --- conversions ---
    k_f32_to_f16<<<blk(N * 128, TB), TB, 0, stream>>>(x, x16, N * 128);
    k_conv_W1<<<blk(R * 192 * 128, TB), TB, 0, stream>>>(W1, W1t, R);
    k_conv_W2<<<blk(R * 64 * 64, TB), TB, 0, stream>>>(W2, W2t, R);

    // --- layer 1: per-relation GAT (H=3, D=64) ---
    for (int r = 0; r < R; ++r) {
        wmma_gemm_f16<4><<<dim3(N / 16, 192 / 64), 128, 0, stream>>>(
            x16, W1t + (size_t)r * 192 * 128, hproj, 128, 192);
        k_attn_logits<<<blk(N * 3, TB), TB, 0, stream>>>(
            hproj, al1 + r * 192, ar1 + r * 192, el, er, N, 3, 64);
        k_fill_zero<<<blk(N * 3, TB), TB, 0, stream>>>(mEnc, N * 3);
        k_fill_zero<<<blk(N * 3, TB), TB, 0, stream>>>((unsigned*)denom, N * 3);
        k_fill_zero<<<blk(N * 192, TB), TB, 0, stream>>>((unsigned*)agg, N * 192);
        k_edge_max<<<blk(E, TB), TB, 0, stream>>>(esrc + (size_t)r * E, edst + (size_t)r * E,
                                                  el, er, mEnc, E, 3);
        k_fixup_m<<<blk(N * 3, TB), TB, 0, stream>>>(mEnc, mDec, N * 3);
        k_edge_exp<<<blk(E, TB), TB, 0, stream>>>(esrc + (size_t)r * E, edst + (size_t)r * E,
                                                  el, er, mDec, pbuf, denom, E, 3);
        k_edge_scatter<<<blk(E * 32, TB), TB, 0, stream>>>(
            esrc + (size_t)r * E, edst + (size_t)r * E, pbuf, denom, hproj, agg, E, 3);
        k_head_mean3<<<blk(N * 64, TB), TB, 0, stream>>>(agg, out1 + (size_t)r * N * 64, N);
    }

    // --- semantic attention 1 + relu -> f16 layer-2 input ---
    k_sem_score<<<R * N, 128, 0, stream>>>(out1, semW1, semb1, semq1, s1);
    k_combine_relu_f16<<<blk(N * 64, TB), TB, 0, stream>>>(out1, s1, hl16, N);

    // --- layer 2: per-relation GAT (H=1, D=64), scatter directly into out2[r] ---
    for (int r = 0; r < R; ++r) {
        wmma_gemm_f16<4><<<dim3(N / 16, 64 / 64), 128, 0, stream>>>(
            hl16, W2t + (size_t)r * 64 * 64, hproj, 64, 64);
        k_attn_logits<<<blk(N, TB), TB, 0, stream>>>(
            hproj, al2 + r * 64, ar2 + r * 64, el, er, N, 1, 64);
        k_fill_zero<<<blk(N, TB), TB, 0, stream>>>(mEnc, N);
        k_fill_zero<<<blk(N, TB), TB, 0, stream>>>((unsigned*)denom, N);
        float* o2r = out2 + (size_t)r * N * 64;
        k_fill_zero<<<blk(N * 64, TB), TB, 0, stream>>>((unsigned*)o2r, N * 64);
        k_edge_max<<<blk(E, TB), TB, 0, stream>>>(esrc + (size_t)r * E, edst + (size_t)r * E,
                                                  el, er, mEnc, E, 1);
        k_fixup_m<<<blk(N, TB), TB, 0, stream>>>(mEnc, mDec, N);
        k_edge_exp<<<blk(E, TB), TB, 0, stream>>>(esrc + (size_t)r * E, edst + (size_t)r * E,
                                                  el, er, mDec, pbuf, denom, E, 1);
        k_edge_scatter<<<blk(E * 32, TB), TB, 0, stream>>>(
            esrc + (size_t)r * E, edst + (size_t)r * E, pbuf, denom, hproj, o2r, E, 1);
    }

    // --- semantic attention 2 -> final output ---
    k_sem_score<<<R * N, 128, 0, stream>>>(out2, semW2, semb2, semq2, s2);
    k_combine_out<<<blk(N * 64, TB), TB, 0, stream>>>(out2, s2, out, N);

    (void)in_sizes; (void)n_in; (void)out_size; (void)ws_size;
}
